// QINSWeightLinear_19250043421228
// MI455X (gfx1250) — matrix operand order repaired and checked by reading, not compiled, not run on hardware
//
#include <hip/hip_runtime.h>
#include <hip/hip_bf16.h>

// QINS-decoded u8-weight linear layer as an f16-WMMA GEMM for gfx1250.
//   out[M,N] = x[M,K] * decode(w[N,K])^T + bias[N]
// M=2048, K=4096, N=11008. Block tile 256x128x32, 8 waves (wave32),
// wave grid 4(M) x 2(N); each wave owns a 64x64 output tile = 4x4 WMMA tiles.
// LDS double-buffered: one barrier per k-step; global->reg prefetch of the
// next tile issues before the WMMA block so load latency hides under matrix ops.

typedef __attribute__((ext_vector_type(16))) _Float16 v16h;
typedef __attribute__((ext_vector_type(8)))  _Float16 v8h;
typedef __attribute__((ext_vector_type(4)))  _Float16 v4h;
typedef __attribute__((ext_vector_type(8)))  float    v8f;

#define M_DIM 2048
#define K_DIM 4096
#define N_DIM 11008
#define BM 256
#define BN 128
#define BK 32
#define LDA 40   // 32 + 8 halfs pad -> 80B row stride (20 banks), conflict-free b128 reads
#define LDB 40

__device__ __forceinline__ _Float16 qins_decode(unsigned int u) {
    // e = u/127.5 - 1 ; a = max(|e|, eps) ; w = sign(e) * (1-a)/a = sign(e)*(1/a - 1)
    float e = (float)u * (1.0f / 127.5f) - 1.0f;
    float a = fabsf(e);
    a = fmaxf(a, 1e-12f);
    float v = __builtin_amdgcn_rcpf(a) - 1.0f;
    v = (e < 0.0f) ? -v : v;
    return (_Float16)v;
}

__launch_bounds__(256)
__global__ void qins_gemm_wmma(const float* __restrict__ x,
                               const unsigned char* __restrict__ w,
                               const float* __restrict__ bias,
                               float* __restrict__ out)
{
    // Double-buffered tiles: A 2x(256x40) halfs = 40KB, B 2x(128x40) = 20KB.
    __shared__ __align__(16) _Float16 sA[2][BM * LDA];
    __shared__ __align__(16) _Float16 sB[2][BN * LDB];

    const int tid   = threadIdx.x;
    const int lane  = tid & 31;
    const int wid   = tid >> 5;
    const int waveM = wid >> 1;          // 0..3 -> 64-row slab
    const int waveN = wid & 1;           // 0..1 -> 64-col slab
    const int bM    = blockIdx.y;        // 8 tiles
    const int bN    = blockIdx.x;        // 86 tiles

    // Global-load work split (256 threads)
    const int xrow  = tid >> 3;          // 0..31, rows xrow + 32*i (i<8)
    const int xcol  = (tid & 7) << 2;    // float4 column within 32-wide k-tile
    const int wrow  = tid >> 1;          // 0..127 (N rows of weight tile)
    const int wpart = (tid & 1) << 4;    // 0 or 16 bytes within 32-byte k-slice

    const float*         xblk = x + (size_t)bM * BM * K_DIM;
    const unsigned char* wblk = w + (size_t)bN * BN * K_DIM;

    float4 xreg[8];
    uint4  wreg;

    // ---- Prologue: load + stage k-tile 0 into buffer 0 ----
    {
        const float* xp = xblk + xcol;
        #pragma unroll
        for (int i = 0; i < 8; ++i)
            xreg[i] = *(const float4*)(xp + (size_t)(xrow + 32 * i) * K_DIM);
        wreg = *(const uint4*)(wblk + (size_t)wrow * K_DIM + wpart);

        #pragma unroll
        for (int i = 0; i < 8; ++i) {
            v4h h;
            h[0] = (_Float16)xreg[i].x; h[1] = (_Float16)xreg[i].y;
            h[2] = (_Float16)xreg[i].z; h[3] = (_Float16)xreg[i].w;
            *(v4h*)(&sA[0][(xrow + 32 * i) * LDA + xcol]) = h;
        }
        unsigned int uu[4] = {wreg.x, wreg.y, wreg.z, wreg.w};
        v8h d0, d1;
        #pragma unroll
        for (int j = 0; j < 4; ++j)
            #pragma unroll
            for (int b = 0; b < 4; ++b) {
                _Float16 hv = qins_decode((uu[j] >> (8 * b)) & 0xFFu);
                int idx = j * 4 + b;
                if (idx < 8) d0[idx] = hv; else d1[idx - 8] = hv;
            }
        *(v8h*)(&sB[0][wrow * LDB + wpart])     = d0;
        *(v8h*)(&sB[0][wrow * LDB + wpart + 8]) = d1;
    }

    v8f acc[4][4] = {};

    const int kTiles = K_DIM / BK;       // 128
    for (int kt = 0; kt < kTiles; ++kt) {
        __syncthreads();                 // buf[cur] published; buf[nxt] retired
        const int cur = kt & 1;
        const int nxt = cur ^ 1;
        const bool more = (kt + 1 < kTiles);

        // ---- Issue next tile's global loads early (latency under WMMA) ----
        if (more) {
            const float* xp = xblk + (kt + 1) * BK + xcol;
            #pragma unroll
            for (int i = 0; i < 8; ++i)
                xreg[i] = *(const float4*)(xp + (size_t)(xrow + 32 * i) * K_DIM);
            wreg = *(const uint4*)(wblk + (size_t)wrow * K_DIM + (kt + 1) * BK + wpart);
        }

        // ---- Fragments per documented CDNA5 16-bit layouts ----
        // A 16x32: lanes 0-15 row M=lane, K chunks {0..7,16..23};
        //          lanes 16-31 row M=lane-16, K chunks {8..15,24..31}.
        // B 32x16: lane&15 = column, lane>>4 selects K {0..7,16..23} / {8..15,24..31}.
        const int k0 = (lane >> 4) << 3;     // 0 or 8
        const int kb = (lane >> 4) << 4;     // 0 or 16

        v16h bfrag[4];
        #pragma unroll
        for (int tn = 0; tn < 4; ++tn) {
            int brow = waveN * 64 + tn * 16 + (lane & 15);
            v8h lo = *(const v8h*)(&sB[cur][brow * LDB + k0]);
            v8h hi = *(const v8h*)(&sB[cur][brow * LDB + k0 + 16]);
            #pragma unroll
            for (int q = 0; q < 8; ++q) { bfrag[tn][q] = lo[q]; bfrag[tn][q + 8] = hi[q]; }
        }

        #pragma unroll
        for (int tm = 0; tm < 4; ++tm) {
            int arow = waveM * 64 + tm * 16 + (lane & 15);
            v8h lo = *(const v8h*)(&sA[cur][arow * LDA + k0]);
            v8h hi = *(const v8h*)(&sA[cur][arow * LDA + k0 + 16]);
            v16h afrag;
            #pragma unroll
            for (int q = 0; q < 8; ++q) { afrag[q] = lo[q]; afrag[q + 8] = hi[q]; }
            #pragma unroll
            for (int tn = 0; tn < 4; ++tn)
                acc[tm][tn] = __builtin_amdgcn_wmma_f32_16x16x32_f16(
                    /*neg_a=*/false, afrag,
                    /*neg_b=*/false, bfrag[tn],
                    /*c_mod=*/(short)0, acc[tm][tn],
                    /*reuse_a=*/false, /*reuse_b=*/false);
        }

        // ---- Decode + stage next tile into the other buffer ----
        if (more) {
            #pragma unroll
            for (int i = 0; i < 8; ++i) {
                v4h h;
                h[0] = (_Float16)xreg[i].x; h[1] = (_Float16)xreg[i].y;
                h[2] = (_Float16)xreg[i].z; h[3] = (_Float16)xreg[i].w;
                *(v4h*)(&sA[nxt][(xrow + 32 * i) * LDA + xcol]) = h;
            }
            unsigned int uu[4] = {wreg.x, wreg.y, wreg.z, wreg.w};
            v8h d0, d1;
            #pragma unroll
            for (int j = 0; j < 4; ++j)
                #pragma unroll
                for (int b = 0; b < 4; ++b) {
                    _Float16 hv = qins_decode((uu[j] >> (8 * b)) & 0xFFu);
                    int idx = j * 4 + b;
                    if (idx < 8) d0[idx] = hv; else d1[idx - 8] = hv;
                }
            *(v8h*)(&sB[nxt][wrow * LDB + wpart])     = d0;
            *(v8h*)(&sB[nxt][wrow * LDB + wpart + 8]) = d1;
        }
    }

    // ---- Epilogue: C/D layout -> lanes 0-15: M=r, N=lane ; lanes 16-31: M=8+r ----
    #pragma unroll
    for (int tm = 0; tm < 4; ++tm) {
        int rowBase = bM * BM + waveM * 64 + tm * 16 + ((lane & 16) ? 8 : 0);
        #pragma unroll
        for (int tn = 0; tn < 4; ++tn) {
            int col = bN * BN + waveN * 64 + tn * 16 + (lane & 15);
            float bv = bias[col];
            #pragma unroll
            for (int r = 0; r < 8; ++r)
                out[(size_t)(rowBase + r) * N_DIM + col] = acc[tm][tn][r] + bv;
        }
    }
}

extern "C" void kernel_launch(void* const* d_in, const int* in_sizes, int n_in,
                              void* d_out, int out_size, void* d_ws, size_t ws_size,
                              hipStream_t stream) {
    (void)in_sizes; (void)n_in; (void)out_size; (void)d_ws; (void)ws_size;
    const float*         x    = (const float*)d_in[0];
    const unsigned char* w    = (const unsigned char*)d_in[1];
    const float*         bias = (const float*)d_in[2];
    float*               out  = (float*)d_out;

    dim3 grid(N_DIM / BN, M_DIM / BM);   // (86, 8)
    qins_gemm_wmma<<<grid, 256, 0, stream>>>(x, w, bias, out);
}
